// ExportModule_66340064854635
// MI455X (gfx1250) — matrix operand order repaired and checked by reading, not compiled, not run on hardware
//
#include <hip/hip_runtime.h>

#define NN 10000
#define EE 320000
#define DD 256
#define LL 5
#define GG 128
#define RW 16

typedef __attribute__((ext_vector_type(16))) __bf16 v16bf;
typedef __attribute__((ext_vector_type(8)))  float  v8f;

struct U8 { unsigned int u[8]; };

__device__ __forceinline__ unsigned short f2bf(float f) {
  unsigned int x = __builtin_bit_cast(unsigned int, f);
  x += 0x7FFFu + ((x >> 16) & 1u);          // round-to-nearest-even
  return (unsigned short)(x >> 16);
}
__device__ __forceinline__ unsigned int pack2(float lo, float hi) {
  return (unsigned int)f2bf(lo) | ((unsigned int)f2bf(hi) << 16);
}

// ---------------- node feature embedding ----------------
// h = w_val[feat%256] + b_val + rwse@w_rwse + b_rwse + deg_emb[clip(deg)]
__global__ void embed_kernel(const int* __restrict__ feat_id,
                             const float* __restrict__ rwse,
                             const int* __restrict__ indeg,
                             const float* __restrict__ w_val,
                             const float* __restrict__ b_val,
                             const float* __restrict__ w_rwse,
                             const float* __restrict__ b_rwse,
                             const float* __restrict__ deg_emb,
                             float* __restrict__ h) {
  int idx = blockIdx.x * 256 + threadIdx.x;     // N*D threads
  int n = idx >> 8, c = idx & 255;
  int hid = feat_id[n] & 255;                   // feat_id >= 0, D=256 pow2
  int dg = indeg[n]; dg = dg < 0 ? 0 : (dg > 1000 ? 1000 : dg);
  float acc = w_val[hid * DD + c] + b_val[c] + b_rwse[c] + deg_emb[dg * DD + c];
  const float* rw = rwse + n * RW;
  #pragma unroll
  for (int k = 0; k < RW; ++k) acc += rw[k] * w_rwse[k * DD + c];
  h[idx] = acc;
}

__global__ void zero_kernel(float* __restrict__ p, int n) {
  int i = blockIdx.x * 256 + threadIdx.x;
  if (i < n) p[i] = 0.0f;
}

// ---------------- edge scatter-add (L2-resident, f32 atomics) ----------------
__global__ void scatter_kernel(const float* __restrict__ h,
                               const int* __restrict__ ei,
                               float* __restrict__ agg) {
  int t = blockIdx.x * 256 + threadIdx.x;       // E*32 threads, exact grid
  int e = t >> 5, c = (t & 31) << 3;            // 32 lanes x 8 floats = 256
  int s = ei[e], d = ei[EE + e];
  const float4* hp = (const float4*)(h + (size_t)s * DD + c);
  float4 v0 = hp[0], v1 = hp[1];
  float* ap = agg + (size_t)d * DD + c;
  atomicAdd(ap + 0, v0.x); atomicAdd(ap + 1, v0.y);
  atomicAdd(ap + 2, v0.z); atomicAdd(ap + 3, v0.w);
  atomicAdd(ap + 4, v1.x); atomicAdd(ap + 5, v1.y);
  atomicAdd(ap + 6, v1.z); atomicAdd(ap + 7, v1.w);
}

// ---------------- weight pre-pack into WMMA B-fragment order ----------------
// Dense B 32x16 bf16 layout (ISA 7.12.2): lane -> col = lane&15, half = lane>>4,
// VGPR j holds K = half*8 + 2j (j<4) or 16 + half*8 + 2(j-4) (j>=4), +1.  wfmt[mat][ct][ks][lane][j] (packed pairs).
__global__ void wfmt_kernel(const float* __restrict__ w1,
                            const float* __restrict__ w2,
                            unsigned int* __restrict__ wfmt) {
  int idx = blockIdx.x * 256 + threadIdx.x;     // 10 * 32768 threads
  int mat = idx >> 15;
  int rem = idx & 32767;
  int l = mat >> 1, m = mat & 1;
  int ct   = rem >> 11;
  int r2   = rem & 2047;
  int ks   = r2 >> 8;
  int r3   = r2 & 255;
  int lane = r3 >> 3, j = r3 & 7;
  int col  = (ct << 4) + (lane & 15);
  int half = lane >> 4;
  int k    = (ks << 5) + (half << 3) + ((j & 4) << 2) + ((j & 3) << 1);
  const float* src = (m == 0 ? w1 : w2) + (size_t)l * DD * DD;
  wfmt[idx] = pack2(src[k * DD + col], src[(k + 1) * DD + col]);
}

// fold BatchNorm into scale/shift
__global__ void bnprep_kernel(const float* __restrict__ g, const float* __restrict__ b,
                              const float* __restrict__ mu, const float* __restrict__ var,
                              float* __restrict__ scale, float* __restrict__ shift) {
  int i = blockIdx.x * 256 + threadIdx.x;
  if (i >= LL * DD) return;
  float s = g[i] * rsqrtf(var[i] + 1e-5f);
  scale[i] = s;
  shift[i] = b[i] - mu[i] * s;
}

// ---------------- 16x16 output tile, K=256 via 8 x wmma_f32_16x16x32_bf16 ----
// A fragments from LDS packed pairs: dense A 16x32 layout -> dwords
// {0..3, 8..11} + half*4 + ks*16 at row*128.
__device__ __forceinline__ v8f gemm_tile(const unsigned int* __restrict__ lds,
                                         const unsigned int* __restrict__ wf,
                                         int ct, int row, int half, int lane) {
  v8f acc = {0.f, 0.f, 0.f, 0.f, 0.f, 0.f, 0.f, 0.f};
  #pragma unroll
  for (int ks = 0; ks < 8; ++ks) {
    const unsigned int* ap = lds + row * 128 + ks * 16 + half * 4;
    uint4 a0 = *(const uint4*)ap;
    uint4 a1 = *(const uint4*)(ap + 8);
    const unsigned int* bp = wf + ((((ct << 3) + ks) << 5) + lane) * 8;
    uint4 b0 = *(const uint4*)bp;
    uint4 b1 = *(const uint4*)(bp + 4);
    U8 au = { { a0.x, a0.y, a0.z, a0.w, a1.x, a1.y, a1.z, a1.w } };
    U8 bu = { { b0.x, b0.y, b0.z, b0.w, b1.x, b1.y, b1.z, b1.w } };
    v16bf av = __builtin_bit_cast(v16bf, au);
    v16bf bv = __builtin_bit_cast(v16bf, bu);
    acc = __builtin_amdgcn_wmma_f32_16x16x32_bf16(false, av, false, bv,
                                                  (short)0, acc, false, false);
  }
  return acc;
}

// ---------------- fused GIN layer MLP: relu(BN(relu(xW1+b1)W2+b2)) -----------
__global__ void __launch_bounds__(256)
gin_mlp_kernel(float* __restrict__ h, const float* __restrict__ agg,
               const unsigned int* __restrict__ wf1,
               const unsigned int* __restrict__ wf2,
               const float* __restrict__ b1, const float* __restrict__ b2,
               const float* __restrict__ bnscale, const float* __restrict__ bnshift,
               const float* __restrict__ epsp) {
  __shared__ __align__(16) unsigned int   xPair[16 * 128];   // 16x256 bf16
  __shared__ __align__(16) unsigned short tHalf[16 * 256];   // 16x256 bf16

  const int tid = threadIdx.x;
  const int rowbase = blockIdx.x << 4;          // 625 blocks * 16 rows = N
  const float ge = 1.0f + epsp[0];

  // phase 0: stage x = agg + (1+eps)*h as bf16 pairs
  {
    int r  = tid >> 4;
    int cb = (tid & 15) << 4;
    size_t off = (size_t)(rowbase + r) * DD + cb;
    const float4* ha = (const float4*)(h + off);
    const float4* aa = (const float4*)(agg + off);
    unsigned int* dst = xPair + r * 128 + (cb >> 1);
    #pragma unroll
    for (int i = 0; i < 4; ++i) {
      float4 a = aa[i], b = ha[i];
      dst[i * 2]     = pack2(a.x + ge * b.x, a.y + ge * b.y);
      dst[i * 2 + 1] = pack2(a.z + ge * b.z, a.w + ge * b.w);
    }
  }
  __syncthreads();

  const int wv = tid >> 5, lane = tid & 31;
  const int row = lane & 15, half = lane >> 4;

  // phase 1: t = relu(x@W1 + b1) -> LDS bf16
  #pragma unroll
  for (int p = 0; p < 2; ++p) {
    int ct = wv + (p << 3);
    v8f acc = gemm_tile(xPair, wf1, ct, row, half, lane);
    int col = (ct << 4) + row;
    float bb = b1[col];
    #pragma unroll
    for (int i = 0; i < 8; ++i) {
      float v = acc[i] + bb;
      v = v > 0.f ? v : 0.f;
      tHalf[(i + (half << 3)) * DD + col] = f2bf(v);   // D layout: M = i + 8*half
    }
  }
  __syncthreads();

  // phase 2: y = t@W2 + b2 ; folded BN ; relu -> h (in place, rows owned by block)
  const unsigned int* tPair = (const unsigned int*)tHalf;
  #pragma unroll
  for (int p = 0; p < 2; ++p) {
    int ct = wv + (p << 3);
    v8f acc = gemm_tile(tPair, wf2, ct, row, half, lane);
    int col = (ct << 4) + row;
    float bb = b2[col], sc = bnscale[col], sh = bnshift[col];
    #pragma unroll
    for (int i = 0; i < 8; ++i) {
      float v = (acc[i] + bb) * sc + sh;
      v = v > 0.f ? v : 0.f;
      h[(size_t)(rowbase + i + (half << 3)) * DD + col] = v;
    }
  }
}

// ---------------- mean pooling per graph ----------------
__global__ void pool_zero_kernel(float* __restrict__ sums, float* __restrict__ cnt) {
  int i = blockIdx.x * 256 + threadIdx.x;
  if (i < GG * DD) sums[i] = 0.0f;
  if (i < GG) cnt[i] = 0.0f;
}
__global__ void pool_accum_kernel(const float* __restrict__ h,
                                  const int* __restrict__ bidx,
                                  float* __restrict__ sums, float* __restrict__ cnt) {
  int idx = blockIdx.x * 256 + threadIdx.x;     // N*D threads
  int n = idx >> 8, c = idx & 255;
  int b = bidx[n];
  atomicAdd(&sums[b * DD + c], h[idx]);
  if (c == 0) atomicAdd(&cnt[b], 1.0f);
}
__global__ void pool_div_kernel(const float* __restrict__ sums,
                                const float* __restrict__ cnt,
                                float* __restrict__ out) {
  int i = blockIdx.x * 256 + threadIdx.x;       // G*D threads
  float c = cnt[i >> 8];
  c = c < 1.f ? 1.f : c;
  out[i] = sums[i] / c;
}

extern "C" void kernel_launch(void* const* d_in, const int* in_sizes, int n_in,
                              void* d_out, int out_size, void* d_ws, size_t ws_size,
                              hipStream_t stream) {
  (void)in_sizes; (void)n_in; (void)out_size; (void)ws_size;
  const int*   feat_id    = (const int*)d_in[0];
  const int*   edge_index = (const int*)d_in[1];
  const int*   batch_idx  = (const int*)d_in[2];
  const float* rwse       = (const float*)d_in[3];
  const int*   in_degrees = (const int*)d_in[4];
  const float* w_val      = (const float*)d_in[5];
  const float* b_val      = (const float*)d_in[6];
  const float* w_rwse     = (const float*)d_in[7];
  const float* b_rwse     = (const float*)d_in[8];
  const float* deg_emb    = (const float*)d_in[9];
  const float* mlp_w1     = (const float*)d_in[10];
  const float* mlp_b1     = (const float*)d_in[11];
  const float* mlp_w2     = (const float*)d_in[12];
  const float* mlp_b2     = (const float*)d_in[13];
  const float* bn_gamma   = (const float*)d_in[14];
  const float* bn_beta    = (const float*)d_in[15];
  const float* bn_mean    = (const float*)d_in[16];
  const float* bn_var     = (const float*)d_in[17];
  const float* eps_gin    = (const float*)d_in[18];

  float* gout = (float*)d_out;                  // [G,D] graph features
  float* h    = gout + GG * DD;                 // [N,D] node features (output 2)

  // workspace layout
  char* ws = (char*)d_ws;
  float*        agg   = (float*)ws;                                   // N*D f32
  unsigned int* wfmt  = (unsigned int*)(ws + (size_t)NN * DD * 4);    // 10*32768 u32
  char* p2 = ws + (size_t)NN * DD * 4 + (size_t)10 * 32768 * 4;
  float* bnscale = (float*)p2;                  // L*D
  float* bnshift = bnscale + LL * DD;           // L*D
  float* sums    = bnshift + LL * DD;           // G*D
  float* cnt     = sums + GG * DD;              // G

  // one-time (per launch, deterministic) prep
  wfmt_kernel<<<(10 * 32768) / 256, 256, 0, stream>>>(mlp_w1, mlp_w2, wfmt);
  bnprep_kernel<<<(LL * DD + 255) / 256, 256, 0, stream>>>(bn_gamma, bn_beta,
                                                           bn_mean, bn_var,
                                                           bnscale, bnshift);
  embed_kernel<<<(NN * DD) / 256, 256, 0, stream>>>(feat_id, rwse, in_degrees,
                                                    w_val, b_val, w_rwse, b_rwse,
                                                    deg_emb, h);

  for (int l = 0; l < LL; ++l) {
    zero_kernel<<<(NN * DD) / 256, 256, 0, stream>>>(agg, NN * DD);
    scatter_kernel<<<(EE * 32) / 256, 256, 0, stream>>>(h, edge_index, agg);
    gin_mlp_kernel<<<NN / 16, 256, 0, stream>>>(
        h, agg,
        wfmt + (size_t)(l * 2 + 0) * 32768,
        wfmt + (size_t)(l * 2 + 1) * 32768,
        mlp_b1 + l * DD, mlp_b2 + l * DD,
        bnscale + l * DD, bnshift + l * DD,
        eps_gin + l);
  }

  pool_zero_kernel<<<(GG * DD) / 256, 256, 0, stream>>>(sums, cnt);
  pool_accum_kernel<<<(NN * DD) / 256, 256, 0, stream>>>(h, batch_idx, sums, cnt);
  pool_div_kernel<<<(GG * DD) / 256, 256, 0, stream>>>(sums, cnt, gout);
}
